// FactorAttnConvRelPosEnc_27805618275136
// MI455X (gfx1250) — matrix operand order repaired and checked by reading, not compiled, not run on hardware
//
#include <hip/hip_runtime.h>
#include <hip/hip_bf16.h>

// ---------------------------------------------------------------------------
// FactorAttnConvRelPosEnc for MI455X (gfx1250, wave32, WMMA).
// fp32 -> fp16 conversion in the LDS tile stage, v_wmma_f32_16x16x32_f16,
// fp32 accumulation. Branchless clamped loads + float4 vector paths +
// register-staged prefetch pipeline.
// ---------------------------------------------------------------------------

typedef __attribute__((ext_vector_type(16))) _Float16 v16h;
typedef __attribute__((ext_vector_type(8)))  float    v8f;

#define LDT    40   // LDS row stride in halves (bank stagger, keeps 4B/8B align)
#define TILE_K 32

union Frag16 { v16h v; unsigned int u[8]; };

// A fragment 16x32 (MxK), ISA 7.12.2 "16-bit A-Matrix 16x32".
__device__ __forceinline__ void load_a_frag(Frag16& f, const _Float16* tile,
                                            int row0, int lane) {
  int m = lane & 15, kg = lane >> 4;
  const _Float16* base = tile + (row0 + m) * LDT;
#pragma unroll
  for (int r = 0; r < 8; ++r) {
    int k0 = (r < 4) ? (8 * kg + 2 * r) : (16 + 8 * kg + 2 * (r - 4));
    f.u[r] = *(const unsigned int*)(base + k0);
  }
}

// B fragment 32x16 (KxN), LDS tile stored transposed as (n,k) row-major:
// lanes 0-15 hold K=0..7,16..23; lanes 16-31 hold K=8..15,24..31 (same as A).
__device__ __forceinline__ void load_b_frag(Frag16& f, const _Float16* tile,
                                            int row0, int lane) {
  int n = lane & 15, kg = lane >> 4;
  const _Float16* base = tile + (row0 + n) * LDT;
#pragma unroll
  for (int r = 0; r < 8; ++r) {
    int k0 = (r < 4) ? (8 * kg + 2 * r) : (16 + 8 * kg + 2 * (r - 4));
    f.u[r] = *(const unsigned int*)(base + k0);
  }
}

// Generic batched, strided GEMM:  C[z][m][n] = alpha * sum_k A[z][m,k]*B[z][n,k] (+bias[n])
//   A element at  A + z*batchA + m*Asm + k*Ask   (fp32 -> fp16 in LDS)
//   B element at  B + z*batchB + n*Bsn + k*Bsk
// MODE 0: C at C + (z/cbInnerN)*cbOuter + (z%cbInnerN)*cbInner + m*ldc + n
// MODE 1: qkv scatter epilogue into separate q/k/v (B,h,N,Ch) arrays.
// AVEC/BVEC: 4 => k-stride==1, float4 global loads (requires K%32==0, stride%4==0).
template <int MODE, int AVEC, int BVEC>
__global__ void __launch_bounds__(256) gemm_f16_wmma(
    const float* __restrict__ A, long batchA, int Asm, int Ask,
    const float* __restrict__ B, long batchB, int Bsn, int Bsk,
    float* __restrict__ C, long cbOuter, long cbInner, int cbInnerN, int ldc,
    const float* __restrict__ bias, float alpha,
    float* __restrict__ qp, float* __restrict__ kp, float* __restrict__ vp,
    int M, int N, int K)
{
  __shared__ __attribute__((aligned(16))) _Float16 As[64 * LDT];
  __shared__ __attribute__((aligned(16))) _Float16 Bs[64 * LDT];

  const int t    = threadIdx.x;
  const int wid  = t >> 5;
  const int lane = t & 31;
  const int m0   = blockIdx.x * 64;
  const int n0   = blockIdx.y * 64;
  const int z    = blockIdx.z;

  const float* Ab = A + (long)z * batchA;
  const float* Bb = B + (long)z * batchB;

  const int tm  = wid & 3;          // wave's 16-row M tile (0..3)
  const int tn0 = (wid >> 2) * 2;   // wave's pair of 16-col N tiles

  v8f acc0 = {};
  v8f acc1 = {};

  // register staging for the global->LDS pipeline
  float4 a4[2], b4[2];
  float  a1[8], b1[8];

  const int ksteps = (K + TILE_K - 1) / TILE_K;

  // ---- branchless clamped fetch of one 64x32 tile into registers ----
  auto fetchA = [&](int k0) {
    if constexpr (AVEC == 4) {
#pragma unroll
      for (int e = 0; e < 2; ++e) {
        int idx = e * 256 + t, row = idx >> 3, vc = idx & 7;
        int gm = m0 + row, gk = k0 + vc * 4;
        const float4* p = (const float4*)(Ab + (long)min(gm, M - 1) * Asm
                                             + min(gk, K - 4));
        float4 val = *p;
        if (!(gm < M && gk < K)) val = make_float4(0.f, 0.f, 0.f, 0.f);
        a4[e] = val;
      }
    } else {
#pragma unroll
      for (int e = 0; e < 8; ++e) {
        int idx = e * 256 + t, row = idx >> 5, col = idx & 31;
        int gm = m0 + row, gk = k0 + col;
        float val = Ab[(long)min(gm, M - 1) * Asm + (long)min(gk, K - 1) * Ask];
        a1[e] = (gm < M && gk < K) ? val : 0.f;
      }
    }
  };
  auto fetchB = [&](int k0) {
    if constexpr (BVEC == 4) {
#pragma unroll
      for (int e = 0; e < 2; ++e) {
        int idx = e * 256 + t, row = idx >> 3, vc = idx & 7;
        int gn = n0 + row, gk = k0 + vc * 4;
        const float4* p = (const float4*)(Bb + (long)min(gn, N - 1) * Bsn
                                             + min(gk, K - 4));
        float4 val = *p;
        if (!(gn < N && gk < K)) val = make_float4(0.f, 0.f, 0.f, 0.f);
        b4[e] = val;
      }
    } else {
#pragma unroll
      for (int e = 0; e < 8; ++e) {
        int idx = e * 256 + t, row = idx >> 5, col = idx & 31;
        int gn = n0 + row, gk = k0 + col;
        float val = Bb[(long)min(gn, N - 1) * Bsn + (long)min(gk, K - 1) * Bsk];
        b1[e] = (gn < N && gk < K) ? val : 0.f;
      }
    }
  };
  auto storeTiles = [&]() {
    if constexpr (AVEC == 4) {
#pragma unroll
      for (int e = 0; e < 2; ++e) {
        int idx = e * 256 + t, row = idx >> 3, vc = idx & 7;
        _Float16* d = &As[row * LDT + vc * 4];
        d[0] = (_Float16)a4[e].x; d[1] = (_Float16)a4[e].y;
        d[2] = (_Float16)a4[e].z; d[3] = (_Float16)a4[e].w;
      }
    } else {
#pragma unroll
      for (int e = 0; e < 8; ++e) {
        int idx = e * 256 + t, row = idx >> 5, col = idx & 31;
        As[row * LDT + col] = (_Float16)a1[e];
      }
    }
    if constexpr (BVEC == 4) {
#pragma unroll
      for (int e = 0; e < 2; ++e) {
        int idx = e * 256 + t, row = idx >> 3, vc = idx & 7;
        _Float16* d = &Bs[row * LDT + vc * 4];
        d[0] = (_Float16)b4[e].x; d[1] = (_Float16)b4[e].y;
        d[2] = (_Float16)b4[e].z; d[3] = (_Float16)b4[e].w;
      }
    } else {
#pragma unroll
      for (int e = 0; e < 8; ++e) {
        int idx = e * 256 + t, row = idx >> 5, col = idx & 31;
        Bs[row * LDT + col] = (_Float16)b1[e];
      }
    }
  };

  fetchA(0);
  fetchB(0);
  for (int kb = 0; kb < ksteps; ++kb) {
    storeTiles();
    __syncthreads();
    if (kb + 1 < ksteps) {        // overlap next-tile HBM loads with the WMMAs
      fetchA((kb + 1) * TILE_K);
      fetchB((kb + 1) * TILE_K);
    }
    Frag16 a, b0f, b1f;
    load_a_frag(a,   As, tm * 16, lane);
    load_b_frag(b0f, Bs, tn0 * 16, lane);
    load_b_frag(b1f, Bs, (tn0 + 1) * 16, lane);
    acc0 = __builtin_amdgcn_wmma_f32_16x16x32_f16(false, a.v, false, b0f.v,
                                                  (short)0, acc0, false, false);
    acc1 = __builtin_amdgcn_wmma_f32_16x16x32_f16(false, a.v, false, b1f.v,
                                                  (short)0, acc1, false, false);
    __syncthreads();
  }

  // C/D layout: lane col = lane&15, row = r + 8*(lane>>4)
  const int cn = lane & 15, rg = lane >> 4;
  const long coff = (MODE == 0)
      ? (long)(z / cbInnerN) * cbOuter + (long)(z % cbInnerN) * cbInner : 0;
#pragma unroll
  for (int r = 0; r < 8; ++r) {
    int gm = m0 + tm * 16 + r + 8 * rg;
    if (gm >= M) continue;
#pragma unroll
    for (int half = 0; half < 2; ++half) {
      int gn = n0 + (tn0 + half) * 16 + cn;
      if (gn >= N) continue;
      float accv = half ? acc1[r] : acc0[r];
      if constexpr (MODE == 1) {
        // gm = b*785+n ; gn = s*512 + h*64 + c  -> q/k/v[(b*8+h)*785+n][c]
        int b = gm / 785, nn = gm - b * 785;
        int s = gn >> 9, rr = gn & 511, hh = rr >> 6, cc = rr & 63;
        float* dst = (s == 0) ? qp : ((s == 1) ? kp : vp);
        dst[(((long)(b * 8 + hh)) * 785 + nn) * 64 + cc] = accv;
      } else {
        float outv = alpha * accv;
        if (bias) outv += bias[gn];
        C[coff + (long)gm * ldc + gn] = outv;
      }
    }
  }
}

// Softmax over token axis N=785 for each of 32*8*64 = 16384 (b,h,c) columns,
// rewritten in place as exp(k-max)/sum. One wave32 per column.
__global__ void __launch_bounds__(256) ksoftmax_kernel(float* __restrict__ k)
{
  const int wid  = threadIdx.x >> 5;
  const int lane = threadIdx.x & 31;
  const int col  = blockIdx.x * 8 + wid;       // 0..16383
  const int bh   = col >> 6, c = col & 63;
  float* p = k + (long)bh * 785 * 64 + c;      // stride 64 between tokens

  float mx = -3.4e38f;
  for (int n = lane; n < 785; n += 32) mx = fmaxf(mx, p[n * 64]);
#pragma unroll
  for (int off = 16; off > 0; off >>= 1) mx = fmaxf(mx, __shfl_xor(mx, off, 32));

  float s = 0.0f;
  for (int n = lane; n < 785; n += 32) s += __expf(p[n * 64] - mx);
#pragma unroll
  for (int off = 16; off > 0; off >>= 1) s += __shfl_xor(s, off, 32);

  const float inv = 1.0f / s;
  for (int n = lane; n < 785; n += 32) p[n * 64] = __expf(p[n * 64] - mx) * inv;
}

// Depthwise conv (3x3/5x5/7x7 per channel split) on v's image tokens,
// multiplied by q and accumulated into attn. One thread per element.
__global__ void __launch_bounds__(256) crpe_kernel(
    const float* __restrict__ q, const float* __restrict__ v,
    float* __restrict__ attn,
    const float* __restrict__ cw3, const float* __restrict__ cb3,
    const float* __restrict__ cw5, const float* __restrict__ cb5,
    const float* __restrict__ cw7, const float* __restrict__ cb7)
{
  int idx = blockIdx.x * 256 + threadIdx.x;   // ((b*784+pos)*8 + h)*64 + c
  int c = idx & 63;
  int r = idx >> 6;
  int h = r & 7;  r >>= 3;
  int pos = r % 784;
  int b   = r / 784;
  if (b >= 32) return;

  int y = pos / 28, x = pos - y * 28;
  int ch = h * 64 + c;
  const float* w; float acc; int ks;
  if (ch < 128)      { ks = 3; w = cw3 + ch * 9;          acc = cb3[ch]; }
  else if (ch < 320) { ks = 5; w = cw5 + (ch - 128) * 25; acc = cb5[ch - 128]; }
  else               { ks = 7; w = cw7 + (ch - 320) * 49; acc = cb7[ch - 320]; }
  int p = ks >> 1;

  const long vbase = ((long)(b * 8 + h)) * 785 * 64;
  for (int ky = 0; ky < ks; ++ky) {
    int yy = y + ky - p;
    if (yy < 0 || yy >= 28) continue;
    for (int kx = 0; kx < ks; ++kx) {
      int xx = x + kx - p;
      if (xx < 0 || xx >= 28) continue;
      acc += w[ky * ks + kx] * v[vbase + (long)(1 + yy * 28 + xx) * 64 + c];
    }
  }
  float qv = q[vbase + (long)(1 + pos) * 64 + c];
  attn[((long)(b * 785 + 1 + pos)) * 512 + ch] += acc * qv;
}

extern "C" void kernel_launch(void* const* d_in, const int* in_sizes, int n_in,
                              void* d_out, int out_size, void* d_ws, size_t ws_size,
                              hipStream_t stream)
{
  (void)in_sizes; (void)n_in; (void)out_size; (void)ws_size;
  const float* x      = (const float*)d_in[0];
  // d_in[1]=H=28, d_in[2]=W=28 (compile-time constants here)
  const float* qkv_w  = (const float*)d_in[3];
  const float* proj_w = (const float*)d_in[4];
  const float* proj_b = (const float*)d_in[5];
  const float* cw3    = (const float*)d_in[6];
  const float* cb3    = (const float*)d_in[7];
  const float* cw5    = (const float*)d_in[8];
  const float* cb5    = (const float*)d_in[9];
  const float* cw7    = (const float*)d_in[10];
  const float* cb7    = (const float*)d_in[11];
  float* out = (float*)d_out;

  // Workspace layout (fp32): q | k (reused as attn) | v | kvT  (~158 MB total)
  const long QE = 32L * 8 * 785 * 64;   // 12,861,440 per tensor
  float* ws = (float*)d_ws;
  float* q    = ws;
  float* k    = q + QE;
  float* v    = k + QE;
  float* kvT  = v + QE;                 // 256 * 64*64, stored [d][c] (transposed)
  float* attn = k;                      // k dead after kv GEMM -> alias

  const dim3 blk(256);
  const int  MT = 32 * 785;             // 25120 token rows

  // 1) qkv = x @ qkv_w^T, scattered to q/k/v in (B,h,N,Ch) layout
  {
    dim3 grid((MT + 63) / 64, (1536 + 63) / 64, 1);
    gemm_f16_wmma<1, 4, 4><<<grid, blk, 0, stream>>>(
        x, 0, 512, 1, qkv_w, 0, 512, 1,
        nullptr, 0, 0, 1, 0, nullptr, 1.0f,
        q, k, v, MT, 1536, 512);
  }

  // 2) softmax over token axis for k (in place, normalized)
  ksoftmax_kernel<<<dim3(16384 / 8), blk, 0, stream>>>(k);

  // 3) kvT[bh][d][c] = sum_n v[n,d] * k_exp[n,c] : batched 64x64, K=785
  //    (transposed output so step 4's B operand is k-contiguous)
  {
    dim3 grid(1, 1, 256);
    gemm_f16_wmma<0, 1, 1><<<grid, blk, 0, stream>>>(
        v, 785L * 64, 1, 64,   k, 785L * 64, 1, 64,
        kvT, 4096, 0, 1, 64, nullptr, 1.0f,
        nullptr, nullptr, nullptr, 64, 64, 785);
  }

  // 4) attn[b,n,h*64+d] = scale * sum_c q[n,c] * kvT[d,c] : batched 785x64, K=64
  {
    dim3 grid((785 + 63) / 64, 1, 256);
    gemm_f16_wmma<0, 4, 4><<<grid, blk, 0, stream>>>(
        q, 785L * 64, 64, 1,   kvT, 4096, 64, 1,
        attn, 785L * 512, 64, 8, 512, nullptr, 0.125f,
        nullptr, nullptr, nullptr, 785, 64, 64);
  }

  // 5) attn += q_img * dwconv(v_img)   (CRPE; cls token row untouched)
  {
    long total = 32L * 8 * 784 * 64;
    crpe_kernel<<<dim3((unsigned)((total + 255) / 256)), blk, 0, stream>>>(
        q, v, attn, cw3, cb3, cw5, cb5, cw7, cb7);
  }

  // 6) out = attn @ proj_w^T + proj_b
  {
    dim3 grid((MT + 63) / 64, (512 + 63) / 64, 1);
    gemm_f16_wmma<0, 4, 4><<<grid, blk, 0, stream>>>(
        attn, 0, 512, 1, proj_w, 0, 512, 1,
        out, 0, 0, 1, 512, proj_b, 1.0f,
        nullptr, nullptr, nullptr, MT, 512, 512);
  }
}
